// MultiHeadAttention_9689446220205
// MI455X (gfx1250) — matrix operand order repaired and checked by reading, not compiled, not run on hardware
//
#include <hip/hip_runtime.h>
#include <hip/hip_bf16.h>

// ---------------------------------------------------------------------------
// MultiHeadAttention for MI455X (gfx1250): f16 WMMA with f32 accumulation.
//   B=2, S=2048, E=1024, H=16, Dh=64
// ---------------------------------------------------------------------------

typedef __attribute__((ext_vector_type(16))) _Float16 v16h;
typedef __attribute__((ext_vector_type(8)))  float    v8f;

#define B_  2
#define S_  2048
#define H_  16
#define D_  64
#define E_  1024

__device__ __forceinline__ v8f wmma_f16(v16h a, v16h b, v8f c) {
  // D = A(16x32 f16) * B(32x16 f16) + C(16x16 f32)
  return __builtin_amdgcn_wmma_f32_16x16x32_f16(
      /*neg_a=*/false, a, /*neg_b=*/false, b,
      /*c_mod=*/(short)0, c, /*reuse_a=*/false, /*reuse_b=*/false);
}

// A-fragment (16x32, f16, row-major source). Per ISA 7.12.2:
// lane = g*16 + r holds row r, K in {g*8..g*8+7} U {16+g*8..16+g*8+7}
// -> two contiguous 16-byte loads per lane.
__device__ __forceinline__ v16h load_a_f16(const _Float16* tile, int ld) {
  int lane = threadIdx.x & 31;
  int r = lane & 15, g = lane >> 4;
  const _Float16* p0 = tile + (size_t)r * ld + g * 8;
  const _Float16* p1 = tile + (size_t)r * ld + 16 + g * 8;
  union { v16h h; uint4 u[2]; } f;
  f.u[0] = *reinterpret_cast<const uint4*>(p0);
  f.u[1] = *reinterpret_cast<const uint4*>(p1);
  return f.h;
}

// Same fragment but converting from f32 source on the fly.
__device__ __forceinline__ v16h load_a_f32cvt(const float* tile, int ld) {
  int lane = threadIdx.x & 31;
  int r = lane & 15, g = lane >> 4;
  const float* p0 = tile + (size_t)r * ld + g * 8;
  const float* p1 = tile + (size_t)r * ld + 16 + g * 8;
  v16h h;
#pragma unroll
  for (int i = 0; i < 8; ++i) {
    h[i]     = (_Float16)p0[i];
    h[8 + i] = (_Float16)p1[i];
  }
  return h;
}

// B-fragment (32x16, f16) from an N-major ("B^T row-major") source:
// element (K=k, N=n) at tileT[n*ld + k]. Lane n holds K in {g*8..g*8+7} U
// {16+g*8..16+g*8+7} -> two contiguous 16-byte loads per lane.
__device__ __forceinline__ v16h load_b_f16(const _Float16* tileT, int ld) {
  int lane = threadIdx.x & 31;
  int n = lane & 15, g = lane >> 4;
  const _Float16* p = tileT + (size_t)n * ld + g * 8;
  union { v16h h; uint4 u[2]; } f;
  f.u[0] = *reinterpret_cast<const uint4*>(p);
  f.u[1] = *reinterpret_cast<const uint4*>(p + 16);
  return f.h;
}

// ---------------------------------------------------------------------------
// Kernel 0: f32 -> f16 weight conversion
// ---------------------------------------------------------------------------
__global__ void mha_cvt_f16_kernel(const float* __restrict__ src,
                                   _Float16* __restrict__ dst, int n) {
  int i = blockIdx.x * blockDim.x + threadIdx.x;
  if (i < n) dst[i] = (_Float16)src[i];
}

// ---------------------------------------------------------------------------
// Kernel 1: per-head QKV projection. One wave per (b,s); the 16 heads are the
// 16 rows of the WMMA tile. out[h,d] = sum_e x[h,e] * W[d,e]  (x @ W^T).
// Q,K stored row-major per (b,h) head: [bh][s][d]; V stored transposed
// [bh][d][s] so the PV B-fragment is a contiguous load.
// ---------------------------------------------------------------------------
__device__ __forceinline__ void proj_one(const float* __restrict__ xrow,
                                         const _Float16* __restrict__ W16,
                                         _Float16* __restrict__ dst,
                                         bool transposed, int b, int s) {
  int lane = threadIdx.x & 31;
  int n = lane & 15, g = lane >> 4;
  v16h a0 = load_a_f32cvt(xrow, D_);        // e = 0..31
  v16h a1 = load_a_f32cvt(xrow + 32, D_);   // e = 32..63
#pragma unroll
  for (int nt = 0; nt < 4; ++nt) {
    v16h b0 = load_b_f16(W16 + nt * 16 * D_, D_);
    v16h b1 = load_b_f16(W16 + nt * 16 * D_ + 32, D_);
    v8f c = {};
    c = wmma_f16(a0, b0, c);
    c = wmma_f16(a1, b1, c);
#pragma unroll
    for (int rr = 0; rr < 8; ++rr) {
      int h = rr + 8 * g;           // C-frag: M = vgpr + 8*lanegroup
      int d = nt * 16 + n;          // C-frag: N = lane%16
      int bh = b * H_ + h;
      size_t idx = transposed ? ((size_t)(bh * D_ + d) * S_ + s)
                              : (((size_t)bh * S_ + s) * D_ + d);
      dst[idx] = (_Float16)c[rr];
    }
  }
}

__global__ __launch_bounds__(128) void mha_proj_kernel(
    const float* __restrict__ q_in, const float* __restrict__ k_in,
    const float* __restrict__ v_in, const _Float16* __restrict__ Wq16,
    const _Float16* __restrict__ Wk16, const _Float16* __restrict__ Wv16,
    _Float16* __restrict__ Q16, _Float16* __restrict__ K16,
    _Float16* __restrict__ Vt16) {
  int bs = blockIdx.x * 4 + (threadIdx.x >> 5);   // 4 waves, one (b,s) each
  int b = bs / S_, s = bs % S_;
  size_t off = (size_t)bs * E_;      // row of [B*S, E]; heads are sub-rows
  proj_one(q_in + off, Wq16, Q16, false, b, s);
  proj_one(k_in + off, Wk16, K16, false, b, s);
  proj_one(v_in + off, Wv16, Vt16, true, b, s);
}

// ---------------------------------------------------------------------------
// Kernel 2: attention. 8 waves cooperate on one (b,h, 16-row q-tile).
// Full score rows (16 x 2048 f32) live in LDS -> exact softmax.
//   Pass 1: 128 k-tiles split 16/wave (QK^T, scale, -> LDS)
//   Softmax: 256 threads, 128 elements each, LDS tree combine
//   Pass 2: 64 PV k-steps split 8/wave; partial ctx reduced through LDS
//           overlaid on the dead score region (keeps LDS ~130 KB ->
//           2 blocks/WGP = 4 waves/SIMD).
// ---------------------------------------------------------------------------
#define SP_ (S_ + 16)  // padded row stride: 16-lane row segments hit
                       // disjoint bank groups (SP_ % 64 == 16)

__global__ __launch_bounds__(256) void mha_attn_kernel(
    const _Float16* __restrict__ Q16, const _Float16* __restrict__ K16,
    const _Float16* __restrict__ Vt16, _Float16* __restrict__ C16) {
  extern __shared__ float smem[];     // 16*SP_ scores | red[256] | sinv[16]
  float* red  = smem + 16 * SP_;
  float* sinv = red + 256;

  int bh = blockIdx.x >> 7;           // / (S_/16)
  int qt = blockIdx.x & 127;
  int wave = threadIdx.x >> 5;
  int lane = threadIdx.x & 31;
  int n = lane & 15, g = lane >> 4;

  // Q tile fragments (every wave loads the same 2 KB tile; L2-resident)
  const _Float16* Qtile = Q16 + (((size_t)bh * S_) + qt * 16) * D_;
  v16h aq0 = load_a_f16(Qtile, D_);   // d = 0..31
  v16h aq1 = load_a_f16(Qtile + 32, D_);

  // ---- Pass 1: scores = (Q K^T) / 8 -> LDS; wave w owns k-tiles [16w,16w+16)
  for (int kt = wave * 16; kt < wave * 16 + 16; ++kt) {
    const _Float16* Ktile = K16 + (((size_t)bh * S_) + kt * 16) * D_;
    __builtin_prefetch(Ktile + 16 * D_, 0, 0);   // global_prefetch next tile
    v16h b0 = load_b_f16(Ktile, D_);
    v16h b1 = load_b_f16(Ktile + 32, D_);
    v8f c = {};
    c = wmma_f16(aq0, b0, c);
    c = wmma_f16(aq1, b1, c);
#pragma unroll
    for (int rr = 0; rr < 8; ++rr)
      smem[(size_t)(rr + 8 * g) * SP_ + kt * 16 + n] = c[rr] * 0.125f;
  }
  __syncthreads();

  // ---- Softmax: thread t -> row t/16, segment t%16, elements seg + 16*i
  {
    int row = threadIdx.x >> 4, seg = threadIdx.x & 15;
    float* p = smem + (size_t)row * SP_;
    float m = -3.402823466e38f;
    for (int i = 0; i < S_ / 16; ++i) m = fmaxf(m, p[seg + 16 * i]);
    red[threadIdx.x] = m;
    __syncthreads();
    float mrow = -3.402823466e38f;
#pragma unroll
    for (int j = 0; j < 16; ++j) mrow = fmaxf(mrow, red[row * 16 + j]);
    __syncthreads();
    float sum = 0.f;
    for (int i = 0; i < S_ / 16; ++i) {
      int idx = seg + 16 * i;
      float e = __expf(p[idx] - mrow);
      p[idx] = e;
      sum += e;
    }
    red[threadIdx.x] = sum;
    __syncthreads();
    float srow = 0.f;
#pragma unroll
    for (int j = 0; j < 16; ++j) srow += red[row * 16 + j];
    if (seg == 0) sinv[row] = 1.0f / srow;
  }
  __syncthreads();

  // ---- Pass 2: ctx partial = P @ V over wave's k-range [8w, 8w+8) of 32
  v8f acc[4] = {};
  for (int kt = wave * 8; kt < wave * 8 + 8; ++kt) {
    v16h ap;
    {
      int r = lane & 15;
      const float* p0 = smem + (size_t)r * SP_ + kt * 32 + g * 8;
      const float* p1 = smem + (size_t)r * SP_ + kt * 32 + 16 + g * 8;
#pragma unroll
      for (int i = 0; i < 8; ++i) {
        ap[i]     = (_Float16)p0[i];
        ap[8 + i] = (_Float16)p1[i];
      }
    }
#pragma unroll
    for (int nt = 0; nt < 4; ++nt) {
      const _Float16* Vtile = Vt16 + ((size_t)bh * D_ + nt * 16) * S_ + kt * 32;
      __builtin_prefetch(Vtile + 32, 0, 0);
      v16h bv = load_b_f16(Vtile, S_);   // V^T: N(=d)-major, contiguous in k
      acc[nt] = wmma_f16(ap, bv, acc[nt]);
    }
  }
  __syncthreads();                       // all waves done reading scores

  // ---- Cross-wave reduction: slabs overlay the dead score region
  float* slabs = smem;                   // 8 x (16x64) f32 = 32 KB
#pragma unroll
  for (int nt = 0; nt < 4; ++nt)
#pragma unroll
    for (int rr = 0; rr < 8; ++rr)
      slabs[wave * 1024 + (rr + 8 * g) * 64 + nt * 16 + n] = acc[nt][rr];
  __syncthreads();

  int b = bh >> 4, h = bh & 15;
#pragma unroll
  for (int i = 0; i < 4; ++i) {
    int e = threadIdx.x + 256 * i;       // 1024 ctx elements
    int row = e >> 6, col = e & 63;
    float s = 0.f;
#pragma unroll
    for (int w = 0; w < 8; ++w) s += slabs[w * 1024 + e];
    float val = s * sinv[row];
    int sq = qt * 16 + row;
    C16[((size_t)(b * S_ + sq)) * E_ + h * D_ + col] = (_Float16)val;
  }
}

// ---------------------------------------------------------------------------
// Kernel 3: out = concat @ Wo^T + bo.  M=4096, N=K=1024.
// 4 waves / block; each wave a 16x64 output tile (128 WMMAs).
// Wo16 (2 MB) and concat (8 MB) both stay resident in the 192 MB L2.
// ---------------------------------------------------------------------------
__global__ __launch_bounds__(128) void mha_outproj_kernel(
    const _Float16* __restrict__ X16, const _Float16* __restrict__ Wo16,
    const float* __restrict__ bo, float* __restrict__ out) {
  int wave = threadIdx.x >> 5;
  int tile = blockIdx.x * 4 + wave;
  int rowtile = tile >> 4;                 // 256 row tiles of 16
  int cg = tile & 15;                      // 16 column groups of 64
  int lane = threadIdx.x & 31;
  int n = lane & 15, g = lane >> 4;

  v8f acc[4] = {};
  const _Float16* Arow = X16 + (size_t)rowtile * 16 * E_;
  for (int ks = 0; ks < E_ / 32; ++ks) {
    v16h a = load_a_f16(Arow + ks * 32, E_);
#pragma unroll
    for (int nt = 0; nt < 4; ++nt) {
      // B element (k, n) = Wo[col, k]: Wo row-major is already N-major.
      const _Float16* Btile = Wo16 + ((size_t)(cg * 64 + nt * 16)) * E_ + ks * 32;
      v16h bb = load_b_f16(Btile, E_);
      acc[nt] = wmma_f16(a, bb, acc[nt]);
    }
  }
#pragma unroll
  for (int nt = 0; nt < 4; ++nt) {
    int col = cg * 64 + nt * 16 + n;
    float bias = bo[col];
#pragma unroll
    for (int rr = 0; rr < 8; ++rr) {
      int row = rowtile * 16 + rr + 8 * g;
      out[(size_t)row * E_ + col] = acc[nt][rr] + bias;
    }
  }
}

// ---------------------------------------------------------------------------
// Host launcher
// ---------------------------------------------------------------------------
extern "C" void kernel_launch(void* const* d_in, const int* in_sizes, int n_in,
                              void* d_out, int out_size, void* d_ws,
                              size_t ws_size, hipStream_t stream) {
  const float* k_in = (const float*)d_in[0];
  const float* q_in = (const float*)d_in[1];
  const float* v_in = (const float*)d_in[2];
  const float* Wq   = (const float*)d_in[3];
  const float* Wk   = (const float*)d_in[4];
  const float* Wv   = (const float*)d_in[5];
  const float* Wo   = (const float*)d_in[6];
  const float* bo   = (const float*)d_in[7];
  float* out = (float*)d_out;

  // Workspace carve-up (all f16)
  char* ws = (char*)d_ws;
  const size_t nHead = (size_t)B_ * H_ * S_ * D_;   // 4 Mi elements
  _Float16* Q16  = (_Float16*)ws; ws += nHead * 2;
  _Float16* K16  = (_Float16*)ws; ws += nHead * 2;
  _Float16* Vt16 = (_Float16*)ws; ws += nHead * 2;
  _Float16* C16  = (_Float16*)ws; ws += (size_t)B_ * S_ * E_ * 2;
  _Float16* Wq16 = (_Float16*)ws; ws += (size_t)D_ * D_ * 2;
  _Float16* Wk16 = (_Float16*)ws; ws += (size_t)D_ * D_ * 2;
  _Float16* Wv16 = (_Float16*)ws; ws += (size_t)D_ * D_ * 2;
  _Float16* Wo16 = (_Float16*)ws; ws += (size_t)E_ * E_ * 2;

  // Stage 0: weight conversion
  mha_cvt_f16_kernel<<<(D_ * D_ + 255) / 256, 256, 0, stream>>>(Wq, Wq16, D_ * D_);
  mha_cvt_f16_kernel<<<(D_ * D_ + 255) / 256, 256, 0, stream>>>(Wk, Wk16, D_ * D_);
  mha_cvt_f16_kernel<<<(D_ * D_ + 255) / 256, 256, 0, stream>>>(Wv, Wv16, D_ * D_);
  mha_cvt_f16_kernel<<<(E_ * E_ + 255) / 256, 256, 0, stream>>>(Wo, Wo16, E_ * E_);

  // Stage 1: QKV projections (4 waves/block, one (b,s) tile per wave)
  mha_proj_kernel<<<B_ * S_ / 4, 128, 0, stream>>>(q_in, k_in, v_in, Wq16,
                                                   Wk16, Wv16, Q16, K16, Vt16);

  // Stage 2: attention (8 cooperating waves per q-tile; ~130 KB dynamic LDS)
  size_t shm = (size_t)(16 * SP_ + 256 + 16) * sizeof(float);
  mha_attn_kernel<<<B_ * H_ * (S_ / 16), 256, shm, stream>>>(Q16, K16, Vt16, C16);

  // Stage 3: output projection + bias
  mha_outproj_kernel<<<(B_ * S_ / 16) * (E_ / 64) / 4, 128, 0, stream>>>(
      C16, Wo16, bo, out);
}
